// GATv2Classifier_12051678233327
// MI455X (gfx1250) — compile-verified, matrix operand order checked by reading, not run on hardware
//
#include <hip/hip_runtime.h>

#define NNODES 50000
#define NEDGES 600000
#define ETOT   (NNODES + NEDGES)   /* 650000 */
#define FIN    128
#define HIDC   64
#define NCLS   10
#define NGR    64
#define NEG_SLOPE 0.2f

typedef __attribute__((ext_vector_type(16))) __bf16 bf16x16;
typedef __attribute__((ext_vector_type(8)))  float  f32x8;

// ---------------------------------------------------------------------------
// bf16 WMMA GEMM: C[M x NCOLS] = A[M x K] * B[K x NCOLS], f32 in memory,
// bf16 fragments, f32 accumulate. One wave owns a 16-row strip and computes
// ALL n-tiles while holding the A fragment in registers (A read exactly once).
// K, NCOLS compile-time -> fully unrolled: K/32 * NCOLS/16 static WMMAs/wave.
// M must be a multiple of 16 (50000 = 3125*16).
// ---------------------------------------------------------------------------
template <int K, int NCOLS>
__global__ void __launch_bounds__(128)
wmma_gemm_bf16(const float* __restrict__ A, const float* __restrict__ B,
               float* __restrict__ C)
{
    constexpr int NT = NCOLS / 16;
    const int lane = threadIdx.x & 31;
    const int wv   = threadIdx.x >> 5;
    const int mt   = blockIdx.x * 4 + wv;          // wave-uniform
    if (mt >= NNODES / 16) return;                 // uniform exit, EXEC stays all-1s
    const int m0   = mt * 16;
    const int hi   = lane >> 4;                    // 0: lanes 0-15, 1: lanes 16-31
    const int col0 = lane & 15;

    f32x8 acc[NT] = {};
    const float* Arow = A + (size_t)(m0 + col0) * K;
#pragma unroll
    for (int k0 = 0; k0 < K; k0 += 32) {
        const int ka0 = k0 + hi * 8;               // elements 0..7  -> K = ka0..ka0+7
        const int ka1 = k0 + 16 + hi * 8;          // elements 8..15 -> K = ka1..ka1+7
        bf16x16 a;
#pragma unroll
        for (int i = 0; i < 8; ++i) a[i]     = (__bf16)Arow[ka0 + i];
#pragma unroll
        for (int i = 0; i < 8; ++i) a[8 + i] = (__bf16)Arow[ka1 + i];
#pragma unroll
        for (int nt = 0; nt < NT; ++nt) {
            const int col = nt * 16 + col0;
            bf16x16 b;
#pragma unroll
            for (int i = 0; i < 8; ++i) b[i]     = (__bf16)B[(size_t)(ka0 + i) * NCOLS + col];
#pragma unroll
            for (int i = 0; i < 8; ++i) b[8 + i] = (__bf16)B[(size_t)(ka1 + i) * NCOLS + col];
            acc[nt] = __builtin_amdgcn_wmma_f32_16x16x32_bf16(
                          false, a, false, b, (short)0, acc[nt], false, false);
        }
    }
#pragma unroll
    for (int nt = 0; nt < NT; ++nt)
#pragma unroll
        for (int r = 0; r < 8; ++r)
            C[(size_t)(m0 + r + 8 * hi) * NCOLS + nt * 16 + col0] = acc[nt][r];
}

// ---------------------------------------------------------------------------
// order-preserving float <-> uint keys for atomicMax-based segment max
// ---------------------------------------------------------------------------
__device__ __forceinline__ unsigned f2sort(float f) {
    unsigned u = __float_as_uint(f);
    return (u & 0x80000000u) ? ~u : (u | 0x80000000u);
}
__device__ __forceinline__ float sort2f(unsigned u) {
    return __uint_as_float((u & 0x80000000u) ? (u & 0x7FFFFFFFu) : ~u);
}

__device__ __forceinline__ void edge_pair(int wid, const long long* ei,
                                          long long& s, long long& d) {
    if (wid < NEDGES) { s = ei[wid]; d = ei[NEDGES + wid]; }
    else              { s = d = (long long)(wid - NEDGES); }   // self loops
}

// ---------------------------------------------------------------------------
// conv1 edge pass (heads=2, ch=64 -> 128 channels). One wave per edge.
// Lanes 0-15 cover head 0, lanes 16-31 head 1 (4 channels per lane).
// ---------------------------------------------------------------------------
__global__ void __launch_bounds__(256)
edge_score_h2(const float* __restrict__ xl, const float* __restrict__ xr,
              const long long* __restrict__ ei, const float* __restrict__ att,
              float* __restrict__ escore, unsigned* __restrict__ emax)
{
    const int wid  = (int)((blockIdx.x * (unsigned)blockDim.x + threadIdx.x) >> 5);
    const int lane = threadIdx.x & 31;
    if (wid >= ETOT) return;
    long long s, d; edge_pair(wid, ei, s, d);
    const float* xls = xl + (size_t)s * 128;
    const float* xrd = xr + (size_t)d * 128;
    float v = 0.f;
#pragma unroll
    for (int i = 0; i < 4; ++i) {
        const int c = lane * 4 + i;
        float m = xls[c] + xrd[c];
        m = m > 0.f ? m : NEG_SLOPE * m;
        v += m * att[c];                 // att flat (2,64): att[c] == att[hd][c&63]
    }
    v += __shfl_xor(v, 8);
    v += __shfl_xor(v, 4);
    v += __shfl_xor(v, 2);
    v += __shfl_xor(v, 1);               // reduces within each 16-lane half
    if ((lane & 15) == 0) {
        const int hd = lane >> 4;
        escore[(size_t)wid * 2 + hd] = v;
        atomicMax(&emax[(size_t)d * 2 + hd], f2sort(v));
    }
}

__global__ void __launch_bounds__(256)
edge_aggr_h2(const float* __restrict__ xl, const long long* __restrict__ ei,
             const float* __restrict__ escore, const unsigned* __restrict__ emax,
             float* __restrict__ denom, float* __restrict__ agg)
{
    const int wid  = (int)((blockIdx.x * (unsigned)blockDim.x + threadIdx.x) >> 5);
    const int lane = threadIdx.x & 31;
    if (wid >= ETOT) return;
    long long s, d; edge_pair(wid, ei, s, d);
    const int hd = lane >> 4;
    const float e  = escore[(size_t)wid * 2 + hd];
    const float mx = sort2f(emax[(size_t)d * 2 + hd]);
    const float p  = __expf(e - mx);
    if ((lane & 15) == 0) atomicAdd(&denom[(size_t)d * 2 + hd], p);
    const float* xls = xl + (size_t)s * 128;
#pragma unroll
    for (int i = 0; i < 4; ++i) {
        const int c = lane * 4 + i;
        atomicAdd(&agg[(size_t)d * 128 + c], p * xls[c]);
    }
}

__global__ void finalize_h2(float* __restrict__ agg, const float* __restrict__ denom,
                            const float* __restrict__ bias)
{
    const long long idx = (long long)blockIdx.x * blockDim.x + threadIdx.x;
    if (idx >= (long long)NNODES * 128) return;
    const int c = (int)(idx & 127);
    const long long i = idx >> 7;
    const int hd = c >> 6;
    float v = agg[idx] / denom[i * 2 + hd] + bias[c];
    agg[idx] = v > 0.f ? v : (__expf(v) - 1.f);      // ELU
}

// ---------------------------------------------------------------------------
// conv2 edge pass (heads=1, ch=64). One wave per edge, 2 channels per lane.
// ---------------------------------------------------------------------------
__global__ void __launch_bounds__(256)
edge_score_h1(const float* __restrict__ xl, const float* __restrict__ xr,
              const long long* __restrict__ ei, const float* __restrict__ att,
              float* __restrict__ escore, unsigned* __restrict__ emax)
{
    const int wid  = (int)((blockIdx.x * (unsigned)blockDim.x + threadIdx.x) >> 5);
    const int lane = threadIdx.x & 31;
    if (wid >= ETOT) return;
    long long s, d; edge_pair(wid, ei, s, d);
    const float* xls = xl + (size_t)s * 64;
    const float* xrd = xr + (size_t)d * 64;
    float v = 0.f;
#pragma unroll
    for (int i = 0; i < 2; ++i) {
        const int c = lane * 2 + i;
        float m = xls[c] + xrd[c];
        m = m > 0.f ? m : NEG_SLOPE * m;
        v += m * att[c];
    }
    v += __shfl_xor(v, 16);
    v += __shfl_xor(v, 8);
    v += __shfl_xor(v, 4);
    v += __shfl_xor(v, 2);
    v += __shfl_xor(v, 1);
    if (lane == 0) {
        escore[wid] = v;
        atomicMax(&emax[d], f2sort(v));
    }
}

__global__ void __launch_bounds__(256)
edge_aggr_h1(const float* __restrict__ xl, const long long* __restrict__ ei,
             const float* __restrict__ escore, const unsigned* __restrict__ emax,
             float* __restrict__ denom, float* __restrict__ agg)
{
    const int wid  = (int)((blockIdx.x * (unsigned)blockDim.x + threadIdx.x) >> 5);
    const int lane = threadIdx.x & 31;
    if (wid >= ETOT) return;
    long long s, d; edge_pair(wid, ei, s, d);
    const float p = __expf(escore[wid] - sort2f(emax[d]));
    if (lane == 0) atomicAdd(&denom[d], p);
    const float* xls = xl + (size_t)s * 64;
#pragma unroll
    for (int i = 0; i < 2; ++i) {
        const int c = lane * 2 + i;
        atomicAdd(&agg[(size_t)d * 64 + c], p * xls[c]);
    }
}

__global__ void finalize_h1(float* __restrict__ agg, const float* __restrict__ denom,
                            const float* __restrict__ bias)
{
    const long long idx = (long long)blockIdx.x * blockDim.x + threadIdx.x;
    if (idx >= (long long)NNODES * 64) return;
    const int c = (int)(idx & 63);
    const long long i = idx >> 6;
    float v = agg[idx] / denom[i] + bias[c];
    agg[idx] = v > 0.f ? v : (__expf(v) - 1.f);      // ELU
}

// ---------------------------------------------------------------------------
// global mean pool + linear head
// ---------------------------------------------------------------------------
__global__ void pool_kernel(const float* __restrict__ h, const long long* __restrict__ batch,
                            float* __restrict__ pooled, float* __restrict__ counts)
{
    const long long idx = (long long)blockIdx.x * blockDim.x + threadIdx.x;
    if (idx >= (long long)NNODES * 64) return;
    const int c = (int)(idx & 63);
    const long long i = idx >> 6;
    const long long g = batch[i];
    atomicAdd(&pooled[g * 64 + c], h[idx]);
    if (c == 0) atomicAdd(&counts[g], 1.f);
}

__global__ void head_kernel(const float* __restrict__ pooled, const float* __restrict__ counts,
                            const float* __restrict__ Wlin, const float* __restrict__ blin,
                            float* __restrict__ out)
{
    const int g = blockIdx.x;
    const int j = threadIdx.x;
    if (j >= NCLS) return;
    const float inv = 1.f / fmaxf(counts[g], 1.f);
    float acc = blin[j];
#pragma unroll
    for (int c = 0; c < 64; ++c)
        acc += pooled[g * 64 + c] * inv * Wlin[c * NCLS + j];
    out[g * NCLS + j] = acc;
}

// ---------------------------------------------------------------------------
extern "C" void kernel_launch(void* const* d_in, const int* in_sizes, int n_in,
                              void* d_out, int out_size, void* d_ws, size_t ws_size,
                              hipStream_t stream)
{
    (void)in_sizes; (void)n_in; (void)out_size; (void)ws_size;
    const float*     x     = (const float*)d_in[0];
    const long long* ei    = (const long long*)d_in[1];
    const long long* batch = (const long long*)d_in[2];
    const float*     Wl1   = (const float*)d_in[3];
    const float*     Wr1   = (const float*)d_in[4];
    const float*     att1  = (const float*)d_in[5];
    const float*     b1    = (const float*)d_in[6];
    const float*     Wl2   = (const float*)d_in[7];
    const float*     Wr2   = (const float*)d_in[8];
    const float*     att2  = (const float*)d_in[9];
    const float*     b2    = (const float*)d_in[10];
    const float*     Wlin  = (const float*)d_in[11];
    const float*     blin  = (const float*)d_in[12];
    float* out = (float*)d_out;

    char* ws = (char*)d_ws;
    size_t off = 0;
    auto alloc = [&](size_t bytes) -> void* {
        off = (off + 255) & ~(size_t)255;
        void* p = ws + off;
        off += bytes;
        return p;
    };
    float*    bufA   = (float*)alloc((size_t)NNODES * 128 * 4);  // xl1 -> xl2
    float*    bufB   = (float*)alloc((size_t)NNODES * 128 * 4);  // xr1 -> xr2
    float*    bufC   = (float*)alloc((size_t)NNODES * 128 * 4);  // agg1->h1, agg2->h2
    float*    escore = (float*)alloc((size_t)ETOT * 2 * 4);
    unsigned* emax   = (unsigned*)alloc((size_t)NNODES * 2 * 4);
    float*    denom  = (float*)alloc((size_t)NNODES * 2 * 4);
    float*    pooled = (float*)alloc((size_t)NGR * HIDC * 4);
    float*    counts = (float*)alloc((size_t)NGR * 4);

    // zero accumulators (key 0 == "below -inf" for the sortable-max)
    hipMemsetAsync(bufC,   0, (size_t)NNODES * 128 * 4, stream);
    hipMemsetAsync(emax,   0, (size_t)NNODES * 2 * 4,   stream);
    hipMemsetAsync(denom,  0, (size_t)NNODES * 2 * 4,   stream);
    hipMemsetAsync(pooled, 0, (size_t)NGR * HIDC * 4,   stream);
    hipMemsetAsync(counts, 0, (size_t)NGR * 4,          stream);

    const int mtiles  = NNODES / 16;                  // 3125
    const int gblocks = (mtiles + 3) / 4;             // 4 waves (m-strips) / block
    const int eblocks = ETOT / 8;                     // 8 edges (waves) / 256-thr block

    // conv1 feature transforms (K=128, 128 out cols)
    wmma_gemm_bf16<128, 128><<<gblocks, 128, 0, stream>>>(x, Wl1, bufA);
    wmma_gemm_bf16<128, 128><<<gblocks, 128, 0, stream>>>(x, Wr1, bufB);

    edge_score_h2<<<eblocks, 256, 0, stream>>>(bufA, bufB, ei, att1, escore, emax);
    edge_aggr_h2 <<<eblocks, 256, 0, stream>>>(bufA, ei, escore, emax, denom, bufC);
    finalize_h2<<<(NNODES * 128 + 255) / 256, 256, 0, stream>>>(bufC, denom, b1);

    // conv2 feature transforms (K=128, 64 out cols); h1 lives in bufC
    wmma_gemm_bf16<128, 64><<<gblocks, 128, 0, stream>>>(bufC, Wl2, bufA);
    wmma_gemm_bf16<128, 64><<<gblocks, 128, 0, stream>>>(bufC, Wr2, bufB);

    hipMemsetAsync(bufC,  0, (size_t)NNODES * 64 * 4, stream);  // agg2
    hipMemsetAsync(emax,  0, (size_t)NNODES * 4,      stream);
    hipMemsetAsync(denom, 0, (size_t)NNODES * 4,      stream);

    edge_score_h1<<<eblocks, 256, 0, stream>>>(bufA, bufB, ei, att2, escore, emax);
    edge_aggr_h1 <<<eblocks, 256, 0, stream>>>(bufA, ei, escore, emax, denom, bufC);
    finalize_h1<<<(NNODES * 64 + 255) / 256, 256, 0, stream>>>(bufC, denom, b2);

    pool_kernel<<<(NNODES * 64 + 255) / 256, 256, 0, stream>>>(bufC, batch, pooled, counts);
    head_kernel<<<NGR, 32, 0, stream>>>(pooled, counts, Wlin, blin, out);
}
